// FeatureWarping_35682588295914
// MI455X (gfx1250) — matrix operand order repaired and checked by reading, not compiled, not run on hardware
//
#include <hip/hip_runtime.h>

// ---------------------------------------------------------------------------
// FeatureWarping (deformable-conv style) for MI455X / gfx1250, wave32.
//  B=4, C=128, H=W=128, K=3, KK=9.  Contraction runs on v_wmma_f32_16x16x32_f16.
// ---------------------------------------------------------------------------

typedef __attribute__((ext_vector_type(16))) _Float16 v16h;
typedef __attribute__((ext_vector_type(8)))  _Float16 v8h;
typedef __attribute__((ext_vector_type(8)))  float    v8f;

#define Bn   4
#define Cn   128
#define Hn   128
#define Wn   128
#define HWn  (Hn * Wn)      // 16384
#define KKn  9
#define RTOT (Cn * KKn)     // 1152 reduction length
#define NCH  36             // 1152 / 32 K-chunks
#define NT   16             // pixel tile (N)
#define SP   1160           // padded LDS row stride (halves): 580 dwords % 64 banks = 4

// ---------------------------------------------------------------------------
// Prep: swizzle deform_w [Cout=128][Cin=128][3][3] (f32) into WMMA A-fragment
// layout (f16).  Reduction index r = k*128 + c.  A-frag (16x32 f16):
//   lane l -> M = l%16 ;  half slot h -> K = (h/8)*16 + (h%8) + 8*(l/16)
// Stored as aw[((otile*36 + chunk)*32 + lane)*16 halves]  (32B per lane, aligned)
// ---------------------------------------------------------------------------
__global__ __launch_bounds__(256)
void fw_prep_weights(const float* __restrict__ dw, _Float16* __restrict__ aw) {
    int gid = blockIdx.x * 256 + threadIdx.x;
    if (gid >= 8 * NCH * 32) return;
    int lane = gid & 31;
    int t    = (gid >> 5) % NCH;
    int ot   = gid / (NCH * 32);
    int m    = lane & 15;
    int g    = lane >> 4;
    int o    = ot * 16 + m;
    v16h v;
#pragma unroll
    for (int h = 0; h < 16; ++h) {
        int K = (h >> 3) * 16 + (h & 7) + 8 * g;
        int r = 32 * t + K;
        int c = r & (Cn - 1);
        int k = r >> 7;                       // r / 128
        v[h] = (_Float16)dw[(o * Cn + c) * KKn + k];
    }
    *(v16h*)(aw + (size_t)((ot * NCH + t) * 32 + lane) * 16) = v;
}

// ---------------------------------------------------------------------------
// Main fused kernel: one workgroup = 16 pixels of one row of one batch.
// ---------------------------------------------------------------------------
__global__ __launch_bounds__(256)
void fw_deform_kernel(const float* __restrict__ feat,
                      const float* __restrict__ flow,
                      const float* __restrict__ offw,   // [18][2][3][3]
                      const float* __restrict__ offb,   // [18]
                      const _Float16* __restrict__ aw,  // prepped A fragments
                      float* __restrict__ out) {
    __shared__ _Float16 S[NT * SP];     // B-panel: S[px][r], r = k*128+c (f16)
    __shared__ int      ci[144 * 4];    // bilinear corner flat indices (per k,px)
    __shared__ float    cw[144 * 4];    // bilinear corner weights (validity folded)

    const int tile = blockIdx.x;                 // 0 .. 4095
    const int x0   = (tile & 7) * NT;
    const int y    = (tile >> 3) & (Hn - 1);
    const int b    = tile >> 10;
    const int tid  = threadIdx.x;

    // ---- Stage 1: offset conv (2->18ch, 3x3, pad 1) + bilinear setup -------
    if (tid < 144) {
        const int kk = tid >> 4;                 // tap 0..8
        const int px = tid & 15;
        const int x  = x0 + px;
        float oy = offb[2 * kk];
        float ox = offb[2 * kk + 1];
#pragma unroll
        for (int ic = 0; ic < 2; ++ic) {
#pragma unroll
            for (int dy = 0; dy < 3; ++dy) {
#pragma unroll
                for (int dx = 0; dx < 3; ++dx) {
                    int yy = y + dy - 1, xx = x + dx - 1;
                    if (yy >= 0 && yy < Hn && xx >= 0 && xx < Wn) {
                        float f = flow[(((size_t)b * 2 + ic) * Hn + yy) * Wn + xx];
                        int wi = ic * 9 + dy * 3 + dx;
                        oy += f * offw[(2 * kk)     * 18 + wi];
                        ox += f * offw[(2 * kk + 1) * 18 + wi];
                    }
                }
            }
        }
        const float py  = (float)y + (float)(kk / 3 - 1) + oy;
        const float pxs = (float)x + (float)(kk % 3 - 1) + ox;
        const float y0f = floorf(py), x0f = floorf(pxs);
        const float wy1 = py - y0f,  wx1 = pxs - x0f;
        const float wy0 = 1.f - wy1, wx0 = 1.f - wx1;
        const int y0i = (int)y0f, x0i = (int)x0f;
#pragma unroll
        for (int corner = 0; corner < 4; ++corner) {
            int dy = corner >> 1, dx = corner & 1;
            int cy = y0i + dy, cx = x0i + dx;
            bool valid = (cy >= 0) && (cy < Hn) && (cx >= 0) && (cx < Wn);
            int yc = min(max(cy, 0), Hn - 1);
            int xc = min(max(cx, 0), Wn - 1);
            ci[tid * 4 + corner] = yc * Wn + xc;
            float wgt = (dy ? wy1 : wy0) * (dx ? wx1 : wx0);
            cw[tid * 4 + corner] = valid ? wgt : 0.f;
        }
    }
    __syncthreads();

    // ---- Stage 2: bilinear gather of all 9 taps x 128 channels -> LDS ------
    {
        const int px = tid & 15;
        const int cb = tid >> 4;                            // 0..15
        const float* fb = feat + (size_t)b * Cn * HWn;
        for (int kk = 0; kk < KKn; ++kk) {
            const int pair = kk * 16 + px;
            const int i0 = ci[pair * 4 + 0], i1 = ci[pair * 4 + 1];
            const int i2 = ci[pair * 4 + 2], i3 = ci[pair * 4 + 3];
            const float w0 = cw[pair * 4 + 0], w1 = cw[pair * 4 + 1];
            const float w2 = cw[pair * 4 + 2], w3 = cw[pair * 4 + 3];
#pragma unroll
            for (int cc = 0; cc < 8; ++cc) {
                const int c = cb + (cc << 4);               // covers 0..127
                const float* fp = fb + (size_t)c * HWn;
                float s = w0 * fp[i0] + w1 * fp[i1] + w2 * fp[i2] + w3 * fp[i3];
                S[px * SP + kk * Cn + c] = (_Float16)s;
            }
        }
    }
    __syncthreads();

    // ---- Stage 3: WMMA contraction  D(16 Cout x 16 px) += A x B ------------
    const int wave = tid >> 5;        // 0..7 -> Cout tile
    const int lane = tid & 31;
    const int pxl  = lane & 15;
    const int grp  = lane >> 4;

    v8f acc = {0.f, 0.f, 0.f, 0.f, 0.f, 0.f, 0.f, 0.f};
    const _Float16* awav = aw + (size_t)(wave * NCH) * 32 * 16;

    for (int t = 0; t < NCH; ++t) {
        // A fragment: one coalesced 32B load per lane (L2-resident weights)
        v16h afr = *(const v16h*)(awav + (size_t)(t * 32 + lane) * 16);
        // B fragment: lane l holds column px=l%16, K = 32t + (h/8)*16 + (h%8) + 8*(l/16)
        const _Float16* sp = S + pxl * SP + 32 * t + 8 * grp;
        v8h lo = *(const v8h*)(sp);
        v8h hi = *(const v8h*)(sp + 16);
        v16h bfr;
#pragma unroll
        for (int i = 0; i < 8; ++i) { bfr[i] = lo[i]; bfr[i + 8] = hi[i]; }
        acc = __builtin_amdgcn_wmma_f32_16x16x32_f16(
            false, afr, false, bfr, (short)0, acc, false, false);
    }

    // ---- Store D: lane l -> N=l%16, rows M = r + 8*(l/16) ------------------
    const int mo = grp * 8;
    float* op = out + ((size_t)b * Cn + (wave * 16 + mo)) * HWn + y * Wn + x0 + pxl;
#pragma unroll
    for (int r = 0; r < 8; ++r) op[(size_t)r * HWn] = acc[r];
}

// ---------------------------------------------------------------------------
extern "C" void kernel_launch(void* const* d_in, const int* in_sizes, int n_in,
                              void* d_out, int out_size, void* d_ws, size_t ws_size,
                              hipStream_t stream) {
    const float* feat = (const float*)d_in[0];   // [4,128,128,128]
    const float* flow = (const float*)d_in[1];   // [4,2,128,128]
    const float* offw = (const float*)d_in[2];   // [18,2,3,3]
    const float* offb = (const float*)d_in[3];   // [18]
    const float* dw   = (const float*)d_in[4];   // [128,128,3,3]
    float*       out  = (float*)d_out;           // [4,128,128,128]
    _Float16*    aw   = (_Float16*)d_ws;         // 8*36*32*32 B = 288 KB

    // Pre-swizzle weights into WMMA A-fragment layout (f16) in workspace.
    fw_prep_weights<<<(8 * NCH * 32 + 255) / 256, 256, 0, stream>>>(dw, aw);

    // Fused offset-conv + bilinear gather + WMMA contraction.
    fw_deform_kernel<<<Bn * Hn * (Wn / NT), 256, 0, stream>>>(
        feat, flow, offw, offb, aw, out);
}
